// Moma_1769526526076
// MI455X (gfx1250) — matrix operand order repaired and checked by reading, not compiled, not run on hardware
//
#include <hip/hip_runtime.h>
#include <hip/hip_bf16.h>
#include <cstdint>

typedef __bf16 bf16;
typedef __attribute__((ext_vector_type(4)))  __bf16 v4bf;
typedef __attribute__((ext_vector_type(8)))  __bf16 v8bf;
typedef __attribute__((ext_vector_type(16))) __bf16 v16bf;
typedef __attribute__((ext_vector_type(8)))  float  v8f;

#define S_SLICES 4
#define BATCH    512
#define MDIM     256
#define BM       64
#define BN       64
#define BK       64
#define LDK      72           // BK + 8 pad: 144B row stride -> conflict-free ds_read_b128
#define TILE_ELEMS ((BM + BN) * LDK)

struct GemmArgs {
  const float* in[6];
  const float* w[6];
  int K[6];
  float* ws;
};

__device__ __forceinline__ v16bf ldfrag(const bf16* p, int o0, int o1) {
  v8bf lo = *(const v8bf*)(p + o0);
  v8bf hi = *(const v8bf*)(p + o1);
  return __builtin_shufflevector(lo, hi, 0,1,2,3,4,5,6,7,8,9,10,11,12,13,14,15);
}

// ---- software-pipeline pieces (guard-free hot path) -------------------------

__device__ __forceinline__ void load_chunk(const float* __restrict__ A,
                                           const float* __restrict__ W, int K,
                                           int row0, int col0, int k0, int t,
                                           float4 aF[8], float4 wF[8]) {
#pragma unroll
  for (int i = 0; i < 8; ++i) {
    int slot = t + i * 128;
    int r    = slot >> 4;           // 0..63
    int kc   = (slot & 15) << 2;    // 0,4,...,60
    aF[i] = *(const float4*)(A + (size_t)(row0 + r) * K + k0 + kc);
    wF[i] = *(const float4*)(W + (size_t)(col0 + r) * K + k0 + kc);
  }
}

__device__ __forceinline__ void store_chunk(bf16* buf, int t,
                                            const float4 aF[8], const float4 wF[8]) {
  bf16* ldsA = buf;
  bf16* ldsB = buf + BM * LDK;
#pragma unroll
  for (int i = 0; i < 8; ++i) {
    int slot = t + i * 128;
    int r    = slot >> 4;
    int kc   = (slot & 15) << 2;
    v4bf pa = { (bf16)aF[i].x, (bf16)aF[i].y, (bf16)aF[i].z, (bf16)aF[i].w };
    v4bf pw = { (bf16)wF[i].x, (bf16)wF[i].y, (bf16)wF[i].z, (bf16)wF[i].w };
    *(v4bf*)(ldsA + r * LDK + kc) = pa;
    *(v4bf*)(ldsB + r * LDK + kc) = pw;
  }
}

// Ragged tail only: per-element guards + zero fill (runs at most once per slice).
__device__ __forceinline__ void stage_tail(const float* __restrict__ A,
                                           const float* __restrict__ W, int K,
                                           int row0, int col0, int k0, int ke,
                                           bf16* buf, int t) {
  bf16* ldsA = buf;
  bf16* ldsB = buf + BM * LDK;
#pragma unroll
  for (int i = 0; i < 8; ++i) {
    int slot = t + i * 128;
    int r    = slot >> 4;
    int kc   = (slot & 15) << 2;
    int gk   = k0 + kc;
    const float* sa = A + (size_t)(row0 + r) * K + gk;
    const float* sw = W + (size_t)(col0 + r) * K + gk;
    float a0 = (gk + 0 < ke) ? sa[0] : 0.f;
    float a1 = (gk + 1 < ke) ? sa[1] : 0.f;
    float a2 = (gk + 2 < ke) ? sa[2] : 0.f;
    float a3 = (gk + 3 < ke) ? sa[3] : 0.f;
    float w0 = (gk + 0 < ke) ? sw[0] : 0.f;
    float w1 = (gk + 1 < ke) ? sw[1] : 0.f;
    float w2 = (gk + 2 < ke) ? sw[2] : 0.f;
    float w3 = (gk + 3 < ke) ? sw[3] : 0.f;
    v4bf pa = { (bf16)a0, (bf16)a1, (bf16)a2, (bf16)a3 };
    v4bf pw = { (bf16)w0, (bf16)w1, (bf16)w2, (bf16)w3 };
    *(v4bf*)(ldsA + r * LDK + kc) = pa;
    *(v4bf*)(ldsB + r * LDK + kc) = pw;
  }
}

struct Frags {
  const bf16 *pa0, *pa1, *pb0, *pb1;
  int half;
};

__device__ __forceinline__ void compute_chunk(const bf16* buf, const Frags& f,
                                              v8f& c00, v8f& c01, v8f& c10, v8f& c11) {
  const bf16* pa0 = buf + (f.pa0 - (const bf16*)0);  // not used; see below
  (void)pa0;
#pragma unroll
  for (int s = 0; s < 2; ++s) {
    // A frag: lane<16 -> K[0..7],K[16..23]; lane>=16 -> K[8..15],K[24..31]
    const int ka = s * 32 + f.half * 8;
    // B frag: lane<16 -> K[0..7],K[16..23]; lane>=16 -> K[8..15],K[24..31]   (col = lane%16)
    const int kb = s * 32 + f.half * 8;
    v16bf a0 = ldfrag(buf + (size_t)(f.pa0 - (const bf16*)0), ka, ka + 16);
    v16bf a1 = ldfrag(buf + (size_t)(f.pa1 - (const bf16*)0), ka, ka + 16);
    v16bf b0 = ldfrag(buf + (size_t)(f.pb0 - (const bf16*)0), kb, kb + 16);
    v16bf b1 = ldfrag(buf + (size_t)(f.pb1 - (const bf16*)0), kb, kb + 16);
    c00 = __builtin_amdgcn_wmma_f32_16x16x32_bf16(false, a0, false, b0, (short)0, c00, false, false);
    c01 = __builtin_amdgcn_wmma_f32_16x16x32_bf16(false, a0, false, b1, (short)0, c01, false, false);
    c10 = __builtin_amdgcn_wmma_f32_16x16x32_bf16(false, a1, false, b0, (short)0, c10, false, false);
    c11 = __builtin_amdgcn_wmma_f32_16x16x32_bf16(false, a1, false, b1, (short)0, c11, false, false);
  }
}

__global__ __launch_bounds__(128) void proj_gemm_kernel(GemmArgs args) {
  __shared__ bf16 lds[2][TILE_ELEMS];   // double buffer: [A(64x72) | B(64x72)]

  const int gemm  = blockIdx.z / S_SLICES;
  const int slice = blockIdx.z % S_SLICES;
  const int K     = args.K[gemm];
  const float* A  = args.in[gemm];
  const float* W  = args.w[gemm];

  const int Kp = (((K + S_SLICES - 1) / S_SLICES) + 31) & ~31;
  const int ks = slice * Kp;
  const int ke = (ks + Kp < K) ? (ks + Kp) : K;
  const int len   = (ke > ks) ? (ke - ks) : 0;
  const int nFull = len / BK;           // guard-free 64-K chunks
  const int tail  = len - nFull * BK;   // 0..63

  const int row0 = blockIdx.x * BM;
  const int col0 = blockIdx.y * BN;
  const int t     = threadIdx.x;
  const int lane  = t & 31;
  const int wid   = t >> 5;
  const int waveM = wid >> 1;
  const int waveN = wid & 1;
  const int half  = lane >> 4;
  const int l16   = lane & 15;

  Frags f;
  f.half = half;
  // store element offsets as "pointer minus null" so compute_chunk can rebase per buffer
  f.pa0 = (const bf16*)0 + (size_t)((waveM * 32 + l16) * LDK);
  f.pa1 = (const bf16*)0 + (size_t)((waveM * 32 + 16 + l16) * LDK);
  f.pb0 = (const bf16*)0 + (size_t)(BM * LDK + (waveN * 32 + l16) * LDK);
  f.pb1 = (const bf16*)0 + (size_t)(BM * LDK + (waveN * 32 + 16 + l16) * LDK);

  v8f c00 = {}, c01 = {}, c10 = {}, c11 = {};

  float4 aF[8], wF[8];
  int buf = 0;
  if (nFull > 0) {
    load_chunk(A, W, K, row0, col0, ks, t, aF, wF);
    store_chunk(lds[0], t, aF, wF);
    __syncthreads();
    // steady state: loads for chunk i+1 issue before compute(i); one barrier/iter
    for (int i = 0; i + 1 < nFull; ++i) {
      load_chunk(A, W, K, row0, col0, ks + (i + 1) * BK, t, aF, wF);
      if (i + 2 < nFull) {   // L2 warm-up two chunks ahead (speculative prefetch)
        int pk = ks + (i + 2) * BK;
        const float* pr = (t < 64) ? (A + (size_t)(row0 + t) * K + pk)
                                   : (W + (size_t)(col0 + (t - 64)) * K + pk);
        __builtin_prefetch(pr, 0, 1);
      }
      compute_chunk(lds[buf], f, c00, c01, c10, c11);
      store_chunk(lds[buf ^ 1], t, aF, wF);
      __syncthreads();
      buf ^= 1;
    }
    compute_chunk(lds[buf], f, c00, c01, c10, c11);
  }
  if (tail > 0) {
    __syncthreads();                    // all reads of lds[] done before overwrite
    stage_tail(A, W, K, row0, col0, ks + nFull * BK, ke, lds[0], t);
    __syncthreads();
    compute_chunk(lds[0], f, c00, c01, c10, c11);
  }

  // C/D layout: lane%16 = N, VGPR v = row (half*8 + v)
  float* dst = args.ws + (size_t)(gemm * S_SLICES + slice) * BATCH * MDIM;
  const int rb = row0 + waveM * 32 + half * 8;
  const int cb = col0 + waveN * 32 + l16;
#pragma unroll
  for (int v = 0; v < 8; ++v) {
    dst[(size_t)(rb +      v) * MDIM + cb     ] = c00[v];
    dst[(size_t)(rb +      v) * MDIM + cb + 16] = c01[v];
    dst[(size_t)(rb + 16 + v) * MDIM + cb     ] = c10[v];
    dst[(size_t)(rb + 16 + v) * MDIM + cb + 16] = c11[v];
  }
}

// One block per batch row: reduce K-slice partials, normalize, fused
// energies + softmaxes + weighted sums + heads. Energies are dots of unit
// 2-vectors (|E|<=1), so exp() needs no max-subtraction.
__global__ __launch_bounds__(256) void fuse_kernel(
    const float* __restrict__ ws,
    const float* __restrict__ W3e, const float* __restrict__ b3e,
    const float* __restrict__ W3m, const float* __restrict__ b3m,
    const float* __restrict__ W3c, const float* __restrict__ b3c,
    float* __restrict__ out) {
  const int b = blockIdx.x;
  const int n = threadIdx.x;

  __shared__ float sex[MDIM], sey[MDIM], smx[MDIM], smy[MDIM], scx[MDIM], scy[MDIM];
  __shared__ float invRowMC[MDIM];
  __shared__ float red[8];

  float v[6];
#pragma unroll
  for (int g = 0; g < 6; ++g) {
    float acc = 0.f;
#pragma unroll
    for (int s = 0; s < S_SLICES; ++s)
      acc += ws[(size_t)(g * S_SLICES + s) * BATCH * MDIM + (size_t)b * MDIM + n];
    v[g] = acc;
  }
  {
    float ie = rsqrtf(v[0]*v[0] + v[1]*v[1]); sex[n] = v[0]*ie; sey[n] = v[1]*ie;
    float im = rsqrtf(v[2]*v[2] + v[3]*v[3]); smx[n] = v[2]*im; smy[n] = v[3]*im;
    float ic = rsqrtf(v[4]*v[4] + v[5]*v[5]); scx[n] = v[4]*ic; scy[n] = v[5]*ic;
  }
  __syncthreads();

  const float ex = sex[n], ey = sey[n];
  const float mx = smx[n], my = smy[n];
  const float cx = scx[n], cy = scy[n];

  // att_cm = softmax(E_mc,-1)^T : w(c, att_cm)[.,n]; s6 = rowsum_n(E_mc)
  float s6 = 0.f, c_cm_x = 0.f, c_cm_y = 0.f;
  for (int j = 0; j < MDIM; ++j) {
    float p = __expf(mx * scx[j] + my * scy[j]);
    s6 += p; c_cm_x += p * scx[j]; c_cm_y += p * scy[j];
  }
  float inv6 = 1.f / s6;
  c_cm_x *= inv6; c_cm_y *= inv6;
  invRowMC[n] = inv6;
  __syncthreads();

  // att_mc = softmax(E_mc,-1) : w(m, att_mc)[.,n]
  float m_mc_x = 0.f, m_mc_y = 0.f;
  for (int i = 0; i < MDIM; ++i) {
    float p = __expf(smx[i] * cx + smy[i] * cy) * invRowMC[i];
    m_mc_x += p * smx[i]; m_mc_y += p * smy[i];
  }

  // att_em = softmax(E_em,1) column softmax: w(e, att_em)[.,n]
  float s1 = 0.f, e_em_x = 0.f, e_em_y = 0.f;
  for (int i = 0; i < MDIM; ++i) {
    float p = __expf(sex[i] * mx + sey[i] * my);
    s1 += p; e_em_x += p * sex[i]; e_em_y += p * sey[i];
  }
  e_em_x /= s1; e_em_y /= s1;

  // att_ec = softmax(E_ec,1): w(e, att_ec)[.,n]
  float s2 = 0.f, e_ec_x = 0.f, e_ec_y = 0.f;
  for (int i = 0; i < MDIM; ++i) {
    float p = __expf(sex[i] * cx + sey[i] * cy);
    s2 += p; e_ec_x += p * sex[i]; e_ec_y += p * sey[i];
  }
  e_ec_x /= s2; e_ec_y /= s2;

  // att_me = softmax(E_em,-1)^T: w(m, att_me)[.,n]
  float s3 = 0.f, m_me_x = 0.f, m_me_y = 0.f;
  for (int j = 0; j < MDIM; ++j) {
    float p = __expf(ex * smx[j] + ey * smy[j]);
    s3 += p; m_me_x += p * smx[j]; m_me_y += p * smy[j];
  }
  m_me_x /= s3; m_me_y /= s3;

  // att_ce = softmax(E_ec,-1)^T: w(c, att_ce)[.,n]
  float s5 = 0.f, c_ce_x = 0.f, c_ce_y = 0.f;
  for (int j = 0; j < MDIM; ++j) {
    float p = __expf(ex * scx[j] + ey * scy[j]);
    s5 += p; c_ce_x += p * scx[j]; c_ce_y += p * scy[j];
  }
  c_ce_x /= s5; c_ce_y /= s5;

  float pe = e_em_x*W3e[n] + e_em_y*W3e[MDIM+n] + e_ec_x*W3e[2*MDIM+n] + e_ec_y*W3e[3*MDIM+n];
  float pm = m_me_x*W3m[n] + m_me_y*W3m[MDIM+n] + m_mc_x*W3m[2*MDIM+n] + m_mc_y*W3m[3*MDIM+n];
  float pc = c_ce_x*W3c[n] + c_ce_y*W3c[MDIM+n] + c_cm_x*W3c[2*MDIM+n] + c_cm_y*W3c[3*MDIM+n];

  float part[3] = { pe, pm, pc };
  const float bias[3] = { b3e[0], b3m[0], b3c[0] };
#pragma unroll
  for (int h = 0; h < 3; ++h) {
    float x = part[h];
#pragma unroll
    for (int off = 16; off > 0; off >>= 1) x += __shfl_down(x, off, 32);
    __syncthreads();
    if ((n & 31) == 0) red[n >> 5] = x;
    __syncthreads();
    if (n == 0) {
      float tot = bias[h];
#pragma unroll
      for (int w8 = 0; w8 < 8; ++w8) tot += red[w8];
      out[h * BATCH + b] = 1.f / (1.f + __expf(-tot));
    }
  }
}

extern "C" void kernel_launch(void* const* d_in, const int* in_sizes, int n_in,
                              void* d_out, int out_size, void* d_ws, size_t ws_size,
                              hipStream_t stream) {
  const float* expr = (const float*)d_in[0];
  const float* mut  = (const float*)d_in[1];
  const float* cna  = (const float*)d_in[2];

  GemmArgs ga;
  ga.in[0] = expr; ga.in[1] = expr;
  ga.in[2] = mut;  ga.in[3] = mut;
  ga.in[4] = cna;  ga.in[5] = cna;
  ga.w[0] = (const float*)d_in[3];
  ga.w[1] = (const float*)d_in[4];
  ga.w[2] = (const float*)d_in[5];
  ga.w[3] = (const float*)d_in[6];
  ga.w[4] = (const float*)d_in[7];
  ga.w[5] = (const float*)d_in[8];
  ga.K[0] = in_sizes[3] / MDIM;
  ga.K[1] = in_sizes[4] / MDIM;
  ga.K[2] = in_sizes[5] / MDIM;
  ga.K[3] = in_sizes[6] / MDIM;
  ga.K[4] = in_sizes[7] / MDIM;
  ga.K[5] = in_sizes[8] / MDIM;
  ga.ws = (float*)d_ws;   // 6*S_SLICES*512*256*4B = 12.6 MB

  dim3 grid(BATCH / BM, MDIM / BN, 6 * S_SLICES);  // 768 blocks x 4 waves
  proj_gemm_kernel<<<grid, 128, 0, stream>>>(ga);

  fuse_kernel<<<BATCH, MDIM, 0, stream>>>(
      (const float*)d_ws,
      (const float*)d_in[9],  (const float*)d_in[10],
      (const float*)d_in[11], (const float*)d_in[12],
      (const float*)d_in[13], (const float*)d_in[14],
      (float*)d_out);
}
